// RegularizedMVOModel_84112639525653
// MI455X (gfx1250) — compile-verified, matrix operand order checked by reading, not run on hardware
//
#include <hip/hip_runtime.h>
#include <hip/hip_bf16.h>

// ---------------------------------------------------------------------------
// Problem constants (match reference)
// ---------------------------------------------------------------------------
#define NA      128      // n assets / features
#define NZ      256      // 2n augmented vars
#define MC      385      // constraint rows: 1 + n + 2n
#define NB      32       // batch
#define ZLD     392      // padded LDS leading dim for Z/Y (>= MC)
#define N_ITERS 500
#define RHO_    0.1f
#define SIGMA_  1e-6f
#define RELAX_  1.6f
#define ALPHA_  0.5f
#define DELTA_  10.0f
#define NEGINF_ (-1e20f)

typedef __attribute__((ext_vector_type(16))) __bf16 v16bf;
typedef __attribute__((ext_vector_type(8)))  float  v8f;
typedef __attribute__((ext_vector_type(4)))  unsigned int v4u;
typedef __attribute__((ext_vector_type(8)))  int    v8i;
typedef __attribute__((ext_vector_type(4)))  int    v4i;

// TDM availability / arity detection:
//  - ROCm 7.2 (clang-22): 5-arg builtin, no gfx1250_TDM.h header
//  - amdgpu-toolchain (clang-23 + therock headers): 6-arg builtin, header present
#if defined(__has_builtin)
#if __has_builtin(__builtin_amdgcn_tensor_load_to_lds)
#define HAVE_TDM 1
#if defined(__has_include)
#if __has_include(<hip/amd_detail/amd_gfx1250_TDM.h>)
#define TDM_6ARG 1
#endif
#endif
#endif
#endif

__device__ __forceinline__ v8f wmma_bf16(v16bf a, v16bf b, v8f c) {
    // 8 args: (neg_a, A, neg_b, B, c_mod, C, reuse_a, reuse_b)
    return __builtin_amdgcn_wmma_f32_16x16x32_bf16(false, a, false, b,
                                                   (short)0, c, false, false);
}

// A/B 16-bit fragment element -> K index (lane half, element e in v16bf)
__device__ __forceinline__ int frag_k(int half, int e) {
    int v = e >> 1, p = e & 1;
    return ((v >> 2) << 4) + (half << 3) + ((v & 3) << 1) + p;
}

__device__ __forceinline__ float dev_exp10(float x) {
    return expf(x * 2.3025850929940457f);
}
__device__ __forceinline__ float dev_sigmoid(float x) {
    return 1.0f / (1.0f + expf(-x));
}

// ---------------------------------------------------------------------------
// Setup kernel 1: y_pred = X @ beta ; P = [-y_pred + p_term, y_pred + p_term]
// ---------------------------------------------------------------------------
__global__ void k_ypred_P(const float* __restrict__ X, const float* __restrict__ beta,
                          const float* __restrict__ thetaE, const float* __restrict__ lg1,
                          float* __restrict__ Pout, float* __restrict__ ypredOut) {
    int f = blockIdx.x * blockDim.x + threadIdx.x;
    if (f >= NB * NA) return;
    int b = f >> 7, j = f & 127;
    float acc = 0.0f;
    for (int k = 0; k < NA; ++k) acc += X[b * NA + k] * beta[k * NA + j];
    ypredOut[f] = acc;
    float g1 = dev_exp10(lg1[0]);
    float pt = ALPHA_ * g1 * dev_sigmoid(thetaE[j]);
    Pout[b * NZ + j]      = -acc + pt;
    Pout[b * NZ + NA + j] =  acc + pt;
}

// ---------------------------------------------------------------------------
// Setup kernel 2: GtG = G^T G
// ---------------------------------------------------------------------------
__global__ void k_gtg(const float* __restrict__ G, float* __restrict__ GtG) {
    for (int f = blockIdx.x * blockDim.x + threadIdx.x; f < NA * NA;
         f += gridDim.x * blockDim.x) {
        int i = f >> 7, j = f & 127;
        float acc = 0.0f;
        for (int k = 0; k < NA; ++k) acc += G[k * NA + i] * G[k * NA + j];
        GtG[f] = acc;
    }
}

// ---------------------------------------------------------------------------
// Setup kernel 3: build augmented [M | I] (256 x 512)
// ---------------------------------------------------------------------------
__global__ void k_build_maug(const float* __restrict__ V, const float* __restrict__ thetaD,
                             const float* __restrict__ lg2, const float* __restrict__ Ar,
                             const float* __restrict__ GtG, float* __restrict__ Maug) {
    for (int f = blockIdx.x * blockDim.x + threadIdx.x; f < NZ * 512;
         f += gridDim.x * blockDim.x) {
        int i = f >> 9, j = f & 511;
        float val;
        if (j < NZ) {
            int ii = i & 127, jj = j & 127;
            float q = DELTA_ * V[ii * NA + jj];
            if (ii == jj) {
                float g2 = dev_exp10(lg2[0]);
                float d  = dev_sigmoid(thetaD[ii]);
                q += (1.0f - ALPHA_) * g2 * d * d;
            }
            float s = ((i < NA) == (j < NA)) ? 1.0f : -1.0f;
            val = s * (q + RHO_ * (Ar[ii] * Ar[jj] + GtG[ii * NA + jj]));
            if (i == j) val += SIGMA_ + RHO_;
        } else {
            val = ((j - 512 + 256) == i) ? 1.0f : 0.0f;  // identity block
        }
        Maug[f] = val;
    }
}

// ---------------------------------------------------------------------------
// Setup kernel 4: Gauss-Jordan inverse of SPD M (one workgroup).
// ---------------------------------------------------------------------------
__global__ void k_gauss_jordan(float* __restrict__ Maug) {
    __shared__ float pivrow[512];
    __shared__ float fac[NZ];
    __shared__ float pivinv_s;
    int t = threadIdx.x;
    for (int k = 0; k < NZ; ++k) {
        if (t == 0) pivinv_s = 1.0f / Maug[k * 512 + k];
        __syncthreads();
        float pinv = pivinv_s;
        for (int j = t; j < 512; j += blockDim.x) {
            float v = Maug[k * 512 + j] * pinv;
            Maug[k * 512 + j] = v;
            pivrow[j] = v;
        }
        for (int i = t; i < NZ; i += blockDim.x) fac[i] = Maug[i * 512 + k];
        __syncthreads();
        for (int idx = t; idx < NZ * 512; idx += blockDim.x) {
            int i = idx >> 9, j = idx & 511;
            if (i != k) Maug[idx] -= fac[i] * pivrow[j];
        }
        __syncthreads();
    }
}

// ---------------------------------------------------------------------------
// Setup kernel 5: pack a KxN matrix into WMMA B-fragment layout, split bf16.
// Fragment storage: frag[(kt*Nt + nt)*512 + lane*16 + e]  (16 bf16 per lane).
// ---------------------------------------------------------------------------
__global__ void k_pack_b(const float* __restrict__ src, int ld, int colOff,
                         int Kt, int Nt, __bf16* __restrict__ hi,
                         __bf16* __restrict__ lo, int transpose) {
    int total = Kt * Nt * 512;
    for (int f = blockIdx.x * blockDim.x + threadIdx.x; f < total;
         f += gridDim.x * blockDim.x) {
        int tile = f >> 9, w = f & 511;
        int l = w >> 4, e = w & 15;
        int kt = tile / Nt, nt = tile % Nt;
        int half = l >> 4, nn = l & 15;
        int k = frag_k(half, e);
        int row = kt * 32 + k, col = nt * 16 + nn;
        float x = transpose ? src[col * ld + row + colOff]
                            : src[row * ld + col + colOff];
        __bf16 h = (__bf16)x;
        hi[f] = h;
        lo[f] = (__bf16)(x - (float)h);
    }
}

// ---------------------------------------------------------------------------
// Persistent ADMM kernel: one workgroup, 8 wave32 waves, all state in LDS.
// P staged into LDS once via the Tensor Data Mover (TDM) when available.
// ---------------------------------------------------------------------------
__global__ __launch_bounds__(256)
void k_admm(const float* __restrict__ Pg,
            const __bf16* __restrict__ MinvHi, const __bf16* __restrict__ MinvLo,
            const __bf16* __restrict__ GHi,    const __bf16* __restrict__ GLo,
            const __bf16* __restrict__ GtHi,   const __bf16* __restrict__ GtLo,
            const float* __restrict__ Ar, const float* __restrict__ bvec,
            const float* __restrict__ hvec, float* __restrict__ out) {
    extern __shared__ char smem_raw[];
    float* sX   = (float*)smem_raw;          // 32x256
    float* sZ   = sX  + NB * NZ;             // 32x392 (385 used)
    float* sY   = sZ  + NB * ZLD;            // 32x392
    float* sRhs = sY  + NB * ZLD;            // 32x256
    float* sXt  = sRhs + NB * NZ;            // 32x256
    float* sU   = sXt + NB * NZ;             // 32x128 (U, later V2)
    float* sP   = sU  + NB * NA;             // 32x256 (P staged via TDM)
    float* sW0  = sP  + NB * NZ;             // 32
    float* sZt0 = sW0 + NB;                  // 32
    float* sLo  = sZt0 + NB;                 // 392
    float* sUp  = sLo + ZLD;                 // 392
    float* sA   = sUp + ZLD;                 // 128
    __bf16* sAhi = (__bf16*)(sA + NA);       // 8192 bf16 (max: 2x8 tiles)
    __bf16* sAlo = sAhi + 2 * 8 * 512;

    const int t = threadIdx.x;
    const int lane = t & 31, wave = t >> 5;

    // ---- stage P: global -> LDS via Tensor Data Mover (one 8192-elem row) ----
#if defined(HAVE_TDM)
    if (wave == 0) {
        unsigned long long ga = (unsigned long long)(size_t)Pg;
        unsigned lds_off = (unsigned)(size_t)sP;
        v4u g0;
        g0.x = 1u;                                        // count=1, user D#
        g0.y = lds_off;                                   // lds_addr
        g0.z = (unsigned)(ga & 0xFFFFFFFFu);              // global_addr lo
        g0.w = (unsigned)((ga >> 32) & 0x01FFFFFFu) | (2u << 30);  // hi | type=2
        v8i g1 = { 0x00020000,    // data_size = 4 bytes
                   0x20000000,    // tensor_dim0 = 8192 (bits 63:48)
                   0x00010000,    // tensor_dim1 = 1
                   0x20000000,    // tile_dim0 = 8192 (bits 127:112)
                   0x00000001,    // tile_dim1 = 1
                   8192,          // tensor_dim0_stride
                   0, 0 };
        v4i gz = {0, 0, 0, 0};
#if defined(TDM_6ARG)
        v8i gz8 = {0, 0, 0, 0, 0, 0, 0, 0};
        __builtin_amdgcn_tensor_load_to_lds(g0, g1, gz, gz, gz8, 0);
#else
        __builtin_amdgcn_tensor_load_to_lds(g0, g1, gz, gz, 0);
#endif
        __builtin_amdgcn_s_wait_tensorcnt(0);
    }
#else
    for (int i = t; i < NB * NZ; i += 256) sP[i] = Pg[i];
#endif

    // ---- init state ----
    for (int i = t; i < NB * NZ; i += 256) { sX[i] = 0.0f; sXt[i] = 0.0f; }
    for (int i = t; i < NB * ZLD; i += 256) { sZ[i] = 0.0f; sY[i] = 0.0f; }
    for (int i = t; i < NA; i += 256) sA[i] = Ar[i];
    for (int i = t; i < MC; i += 256) {
        sLo[i] = (i == 0) ? bvec[0] : NEGINF_;
        sUp[i] = (i == 0) ? bvec[0] : ((i <= NA) ? hvec[i - 1] : 0.0f);
    }
    __syncthreads();

    for (int it = 0; it < N_ITERS; ++it) {
        // phase 1: pack Wg = rho*Z[:,1:129]-Y[:,1:129] into A-frags; W0 column
        for (int f = t; f < 2 * 4 * 512; f += 256) {
            int tile = f >> 9, w = f & 511;
            int l = w >> 4, e = w & 15;
            int mt = tile >> 2, kt = tile & 3;
            int half = l >> 4, mm = l & 15;
            int k = frag_k(half, e);
            int m = mt * 16 + mm, kk = kt * 32 + k;
            float wv = RHO_ * sZ[m * ZLD + 1 + kk] - sY[m * ZLD + 1 + kk];
            __bf16 hv = (__bf16)wv;
            sAhi[f] = hv;
            sAlo[f] = (__bf16)(wv - (float)hv);
        }
        if (t < NB) sW0[t] = RHO_ * sZ[t * ZLD] - sY[t * ZLD];
        __syncthreads();

        // phase 2: GEMM1  U(32x128) = Wg @ G   (16 tiles, 2 per wave)
        for (int tt = wave * 2; tt < wave * 2 + 2; ++tt) {
            int mt = tt >> 3, nt = tt & 7;
            v8f acc = {};
            for (int kt = 0; kt < 4; ++kt) {
                v16bf ahi = *(const v16bf*)&sAhi[(mt * 4 + kt) * 512 + lane * 16];
                v16bf alo = *(const v16bf*)&sAlo[(mt * 4 + kt) * 512 + lane * 16];
                v16bf bhi = *(const v16bf*)&GHi[(kt * 8 + nt) * 512 + lane * 16];
                v16bf blo = *(const v16bf*)&GLo[(kt * 8 + nt) * 512 + lane * 16];
                acc = wmma_bf16(ahi, bhi, acc);
                acc = wmma_bf16(ahi, blo, acc);
                acc = wmma_bf16(alo, bhi, acc);
            }
            int nn = nt * 16 + (lane & 15);
            int mb = mt * 16 + (lane >> 4) * 8;
#pragma unroll
            for (int r = 0; r < 8; ++r) sU[(mb + r) * NA + nn] = acc[r];
        }
        __syncthreads();

        // phase 3a: rhs = sigma*X - P + C^T w (structured)
        for (int f = t; f < NB * NZ; f += 256) {
            int m = f >> 8, i = f & 255;
            int ii = i & 127;
            float sgn = (i < NA) ? 1.0f : -1.0f;
            float wt = RHO_ * sZ[m * ZLD + 129 + i] - sY[m * ZLD + 129 + i];
            sRhs[f] = SIGMA_ * sX[f] - sP[f] +
                      sgn * (sA[ii] * sW0[m] + sU[m * NA + ii]) - wt;
        }
        __syncthreads();
        // phase 3b: pack rhs into A-frags (2 mtiles x 8 ktiles)
        for (int f = t; f < 2 * 8 * 512; f += 256) {
            int tile = f >> 9, w = f & 511;
            int l = w >> 4, e = w & 15;
            int mt = tile >> 3, kt = tile & 7;
            int half = l >> 4, mm = l & 15;
            int k = frag_k(half, e);
            float x = sRhs[(mt * 16 + mm) * NZ + kt * 32 + k];
            __bf16 hv = (__bf16)x;
            sAhi[f] = hv;
            sAlo[f] = (__bf16)(x - (float)hv);
        }
        __syncthreads();

        // phase 4: GEMM2  Xt(32x256) = rhs @ Minv  (32 tiles, 4 per wave)
        for (int tt = wave * 4; tt < wave * 4 + 4; ++tt) {
            int mt = tt >> 4, nt = tt & 15;
            v8f acc = {};
            for (int kt = 0; kt < 8; ++kt) {
                v16bf ahi = *(const v16bf*)&sAhi[(mt * 8 + kt) * 512 + lane * 16];
                v16bf alo = *(const v16bf*)&sAlo[(mt * 8 + kt) * 512 + lane * 16];
                v16bf bhi = *(const v16bf*)&MinvHi[(kt * 16 + nt) * 512 + lane * 16];
                v16bf blo = *(const v16bf*)&MinvLo[(kt * 16 + nt) * 512 + lane * 16];
                acc = wmma_bf16(ahi, bhi, acc);
                acc = wmma_bf16(ahi, blo, acc);
                acc = wmma_bf16(alo, bhi, acc);
            }
            int nn = nt * 16 + (lane & 15);
            int mb = mt * 16 + (lane >> 4) * 8;
#pragma unroll
            for (int r = 0; r < 8; ++r) sXt[(mb + r) * NZ + nn] = acc[r];
        }
        __syncthreads();

        // phase 5: pack Xd = Xt[:, :128]-Xt[:, 128:] into A-frags; Zt0 = Xd@A^T
        for (int f = t; f < 2 * 4 * 512; f += 256) {
            int tile = f >> 9, w = f & 511;
            int l = w >> 4, e = w & 15;
            int mt = tile >> 2, kt = tile & 3;
            int half = l >> 4, mm = l & 15;
            int k = frag_k(half, e);
            int m = mt * 16 + mm, kk = kt * 32 + k;
            float x = sXt[m * NZ + kk] - sXt[m * NZ + NA + kk];
            __bf16 hv = (__bf16)x;
            sAhi[f] = hv;
            sAlo[f] = (__bf16)(x - (float)hv);
        }
        if (t < NB) {
            float a0 = 0.0f;
            for (int i = 0; i < NA; ++i)
                a0 += sA[i] * (sXt[t * NZ + i] - sXt[t * NZ + NA + i]);
            sZt0[t] = a0;
        }
        __syncthreads();

        // phase 6: GEMM3  V2(32x128) = Xd @ G^T  -> reuse sU
        for (int tt = wave * 2; tt < wave * 2 + 2; ++tt) {
            int mt = tt >> 3, nt = tt & 7;
            v8f acc = {};
            for (int kt = 0; kt < 4; ++kt) {
                v16bf ahi = *(const v16bf*)&sAhi[(mt * 4 + kt) * 512 + lane * 16];
                v16bf alo = *(const v16bf*)&sAlo[(mt * 4 + kt) * 512 + lane * 16];
                v16bf bhi = *(const v16bf*)&GtHi[(kt * 8 + nt) * 512 + lane * 16];
                v16bf blo = *(const v16bf*)&GtLo[(kt * 8 + nt) * 512 + lane * 16];
                acc = wmma_bf16(ahi, bhi, acc);
                acc = wmma_bf16(ahi, blo, acc);
                acc = wmma_bf16(alo, bhi, acc);
            }
            int nn = nt * 16 + (lane & 15);
            int mb = mt * 16 + (lane >> 4) * 8;
#pragma unroll
            for (int r = 0; r < 8; ++r) sU[(mb + r) * NA + nn] = acc[r];
        }
        __syncthreads();

        // phase 7: relax/clip/dual updates
        for (int f = t; f < NB * NZ; f += 256)
            sX[f] = RELAX_ * sXt[f] + (1.0f - RELAX_) * sX[f];
        for (int f = t; f < NB * MC; f += 256) {
            int m = f / MC, c = f - m * MC;
            float zt;
            if (c == 0)        zt = sZt0[m];
            else if (c <= NA)  zt = sU[m * NA + (c - 1)];
            else               zt = -sXt[m * NZ + (c - 129)];
            float zo = sZ[m * ZLD + c], yo = sY[m * ZLD + c];
            float zr = RELAX_ * zt + (1.0f - RELAX_) * zo;
            float zn = zr + yo * (1.0f / RHO_);
            zn = fminf(fmaxf(zn, sLo[c]), sUp[c]);
            sY[m * ZLD + c] = yo + RHO_ * (zr - zn);
            sZ[m * ZLD + c] = zn;
        }
        __syncthreads();
    }

    // output: z_batch = u[:, :n] - u[:, n:]
    for (int f = t; f < NB * NA; f += 256) {
        int m = f >> 7, j = f & 127;
        out[f] = sX[m * NZ + j] - sX[m * NZ + NA + j];
    }
}

// ---------------------------------------------------------------------------
// Launch
// ---------------------------------------------------------------------------
extern "C" void kernel_launch(void* const* d_in, const int* in_sizes, int n_in,
                              void* d_out, int out_size, void* d_ws, size_t ws_size,
                              hipStream_t stream) {
    (void)in_sizes; (void)n_in; (void)out_size; (void)ws_size;
    const float* X    = (const float*)d_in[0];
    const float* V    = (const float*)d_in[1];
    const float* beta = (const float*)d_in[2];
    const float* thE  = (const float*)d_in[3];
    const float* thD  = (const float*)d_in[4];
    const float* lg1  = (const float*)d_in[5];
    const float* lg2  = (const float*)d_in[6];
    const float* Ar   = (const float*)d_in[7];
    const float* bv   = (const float*)d_in[8];
    const float* G    = (const float*)d_in[9];
    const float* hv   = (const float*)d_in[10];
    float* out = (float*)d_out;  // [0..4095]=z_batch, [4096..8191]=y_pred

    char* ws = (char*)d_ws;
    float*  Maug   = (float*)ws;  ws += (size_t)NZ * 512 * 4;
    float*  GtG    = (float*)ws;  ws += (size_t)NA * NA * 4;
    float*  P      = (float*)ws;  ws += (size_t)NB * NZ * 4;
    __bf16* MinvHi = (__bf16*)ws; ws += (size_t)8 * 16 * 512 * 2;
    __bf16* MinvLo = (__bf16*)ws; ws += (size_t)8 * 16 * 512 * 2;
    __bf16* GHi    = (__bf16*)ws; ws += (size_t)4 * 8 * 512 * 2;
    __bf16* GLo    = (__bf16*)ws; ws += (size_t)4 * 8 * 512 * 2;
    __bf16* GtHi   = (__bf16*)ws; ws += (size_t)4 * 8 * 512 * 2;
    __bf16* GtLo   = (__bf16*)ws; ws += (size_t)4 * 8 * 512 * 2;

    k_ypred_P<<<(NB * NA + 255) / 256, 256, 0, stream>>>(X, beta, thE, lg1, P, out + NB * NA);
    k_gtg<<<64, 256, 0, stream>>>(G, GtG);
    k_build_maug<<<512, 256, 0, stream>>>(V, thD, lg2, Ar, GtG, Maug);
    k_gauss_jordan<<<1, 1024, 0, stream>>>(Maug);
    k_pack_b<<<128, 256, 0, stream>>>(Maug, 512, 256, 8, 16, MinvHi, MinvLo, 0);
    k_pack_b<<<64, 256, 0, stream>>>(G, NA, 0, 4, 8, GHi, GLo, 0);
    k_pack_b<<<64, 256, 0, stream>>>(G, NA, 0, 4, 8, GtHi, GtLo, 1);

    size_t smem = (size_t)(NB * NZ        // sX
                 + NB * ZLD + NB * ZLD    // sZ, sY
                 + NB * NZ + NB * NZ      // sRhs, sXt
                 + NB * NA                // sU
                 + NB * NZ                // sP (TDM-staged)
                 + NB + NB                // sW0, sZt0
                 + ZLD + ZLD              // sLo, sUp
                 + NA) * sizeof(float)    // sA
                 + (size_t)2 * 8 * 512 * 2 * sizeof(__bf16);  // sAhi, sAlo
    k_admm<<<1, 256, smem, stream>>>(P, MinvHi, MinvLo, GHi, GLo, GtHi, GtLo,
                                     Ar, bv, hv, out);
}
